// GraphTransformer_4982162063452
// MI455X (gfx1250) — matrix operand order repaired and checked by reading, not compiled, not run on hardware
//
#include <hip/hip_runtime.h>
#include <hip/hip_bf16.h>

// ---------------------------------------------------------------------------
// Problem constants (match reference)
// ---------------------------------------------------------------------------
#define NN    20000
#define EE    200000
#define INF_  512
#define HH    4
#define CC    64
#define EDIM_ 384
#define GG    64
#define NC_   751
#define ETOT  (EE + NN)      // edges + self loops
#define NEG_GATV2 0.2f
#define NEG_ACT   0.01f

// ---------------------------------------------------------------------------
// Plain ext-vector types (trivially copyable -> union-safe, unlike HIP float4)
// ---------------------------------------------------------------------------
typedef __attribute__((ext_vector_type(2)))  float    f32x2;
typedef __attribute__((ext_vector_type(4)))  float    f32x4;
typedef __attribute__((ext_vector_type(4)))  unsigned u32x4;
typedef __attribute__((ext_vector_type(8)))  float    v8f;
typedef __attribute__((ext_vector_type(16))) __bf16   v16bf;

union Frag { u32x4 q[2]; v16bf v; };

// fp32 -> bf16 (round to nearest even), bit-level so no scalar __bf16 math
__device__ __forceinline__ unsigned short f2bf(float f) {
    unsigned u = __float_as_uint(f);
    u += 0x7FFFu + ((u >> 16) & 1u);
    return (unsigned short)(u >> 16);
}
__device__ __forceinline__ unsigned pack2(float a, float b) {
    return (unsigned)f2bf(a) | ((unsigned)f2bf(b) << 16);
}
__device__ __forceinline__ float lrelu(float x, float s) { return x > 0.f ? x : s * x; }

// order-preserving float<->uint encoding for atomic max (enc(0x0)=valid -inf-ish floor)
__device__ __forceinline__ unsigned encf(float f) {
    unsigned u = __float_as_uint(f);
    return (u >> 31) ? ~u : (u | 0x80000000u);
}
__device__ __forceinline__ float decf(unsigned e) {
    return (e >> 31) ? __uint_as_float(e & 0x7FFFFFFFu) : __uint_as_float(~e);
}

// ---------------------------------------------------------------------------
// fill
// ---------------------------------------------------------------------------
__global__ void k_fill_u32(unsigned* p, unsigned v, size_t n) {
    size_t i = (size_t)blockIdx.x * blockDim.x + threadIdx.x;
    size_t stride = (size_t)gridDim.x * blockDim.x;
    for (; i < n; i += stride) p[i] = v;
}

// ---------------------------------------------------------------------------
// self-loop attr: scatter sum + count over col = edge_index[1]
// ---------------------------------------------------------------------------
__global__ void k_scatter_attr(const float* __restrict__ ea, const int* __restrict__ ei,
                               float* __restrict__ lsum, float* __restrict__ cnt) {
    size_t t = (size_t)blockIdx.x * blockDim.x + threadIdx.x;
    if (t >= (size_t)EE * EDIM_) return;
    int e = (int)(t / EDIM_);
    int c = (int)(t % EDIM_);
    int col = ei[EE + e];
    atomicAdd(&lsum[(size_t)col * EDIM_ + c], ea[t]);
    if (c == 0) atomicAdd(&cnt[col], 1.0f);
}
__global__ void k_loop_div(float* __restrict__ lattr, const float* __restrict__ cnt) {
    size_t t = (size_t)blockIdx.x * blockDim.x + threadIdx.x;
    if (t >= (size_t)NN * EDIM_) return;
    lattr[t] = lattr[t] / fmaxf(cnt[t / EDIM_], 1.0f);
}

// ---------------------------------------------------------------------------
// Generic bf16 WMMA GEMM: out[M x NCOL] = A[M x K] @ B[K x NCOL] (+bias)
// A rows >= Asplit come from A1 (virtual concat for self-loop attrs).
// Block: 256 thr (8 waves), 64x64 output tile, K-step 32.
// Wave w: row-tile (w>>1)*16, col half (w&1)*32 -> two 16x16 accumulators.
// ---------------------------------------------------------------------------
template <int K, int NCOL>
__global__ __launch_bounds__(256) void k_gemm_bf16(
    const float* __restrict__ A0, const float* __restrict__ A1, int Asplit, int M,
    const float* __restrict__ B, const float* __restrict__ bias,
    float* __restrict__ out) {
    __shared__ __align__(16) unsigned short As[64][40];  // padded stride: 80B
    __shared__ __align__(16) unsigned short Bs[32][72];  // padded stride: 144B

    const int tid  = threadIdx.x;
    const int lane = tid & 31;
    const int wv   = tid >> 5;
    const int mt   = wv >> 1;         // row-tile in block: 0..3
    const int ct2  = wv & 1;          // col half: 0..1
    const int lm16 = lane & 15;
    const int lg   = lane >> 4;

    const int blockM  = blockIdx.y * 64;
    const int colBase = blockIdx.x * 64;

    v8f acc0 = {0.f,0.f,0.f,0.f,0.f,0.f,0.f,0.f};
    v8f acc1 = {0.f,0.f,0.f,0.f,0.f,0.f,0.f,0.f};

    for (int k0 = 0; k0 < K; k0 += 32) {
        // ---- stage A tile 64x32 fp32 -> bf16 LDS (8 floats per thread) ----
        {
            int row = tid >> 2;
            int col = (tid & 3) * 8;
            int gr  = blockM + row;
            f32x4 f0 = {0.f,0.f,0.f,0.f}, f1 = {0.f,0.f,0.f,0.f};
            if (gr < M) {
                const float* src = (gr < Asplit)
                    ? (A0 + (size_t)gr * K) : (A1 + (size_t)(gr - Asplit) * K);
                src += k0 + col;
                f0 = *(const f32x4*)(src);
                f1 = *(const f32x4*)(src + 4);
            }
            u32x4 p = { pack2(f0.x, f0.y), pack2(f0.z, f0.w),
                        pack2(f1.x, f1.y), pack2(f1.z, f1.w) };
            *(u32x4*)&As[row][col] = p;
        }
        // ---- stage B tile 32x64 fp32 -> bf16 LDS ----
        {
            int row = tid >> 3;
            int col = (tid & 7) * 8;
            const float* src = B + (size_t)(k0 + row) * NCOL + colBase + col;
            f32x4 f0 = *(const f32x4*)(src);
            f32x4 f1 = *(const f32x4*)(src + 4);
            u32x4 p = { pack2(f0.x, f0.y), pack2(f0.z, f0.w),
                        pack2(f1.x, f1.y), pack2(f1.z, f1.w) };
            *(u32x4*)&Bs[row][col] = p;
        }
        __syncthreads();

        // ---- fragments (ds_load_b128) + WMMA ----
        Frag af, b0, b1;
        af.q[0] = *(const u32x4*)&As[mt * 16 + lm16][8 * lg];
        af.q[1] = *(const u32x4*)&As[mt * 16 + lm16][16 + 8 * lg];
        auto bq = [&](int kb, int n) -> u32x4 {
            return u32x4{ (unsigned)Bs[kb + 0][n] | ((unsigned)Bs[kb + 1][n] << 16),
                          (unsigned)Bs[kb + 2][n] | ((unsigned)Bs[kb + 3][n] << 16),
                          (unsigned)Bs[kb + 4][n] | ((unsigned)Bs[kb + 5][n] << 16),
                          (unsigned)Bs[kb + 6][n] | ((unsigned)Bs[kb + 7][n] << 16) };
        };
        b0.q[0] = bq(8 * lg, ct2 * 32 + lm16);
        b0.q[1] = bq(16 + 8 * lg, ct2 * 32 + lm16);
        b1.q[0] = bq(8 * lg, ct2 * 32 + 16 + lm16);
        b1.q[1] = bq(16 + 8 * lg, ct2 * 32 + 16 + lm16);

        acc0 = __builtin_amdgcn_wmma_f32_16x16x32_bf16(
                   false, af.v, false, b0.v, (short)0, acc0, false, false);
        acc1 = __builtin_amdgcn_wmma_f32_16x16x32_bf16(
                   false, af.v, false, b1.v, (short)0, acc1, false, false);
        __syncthreads();
    }

    // ---- epilogue: D layout (lanes 0-15: M=vgpr, lanes 16-31: M=vgpr+8) ----
    const int lm = (lane >> 4) * 8;
    const int ln = lane & 15;
#pragma unroll
    for (int t = 0; t < 2; ++t) {
        const v8f& a = t ? acc1 : acc0;
        int gn = colBase + ct2 * 32 + t * 16 + ln;
        float bv = bias ? bias[gn] : 0.0f;
#pragma unroll
        for (int v = 0; v < 8; ++v) {
            int gm = blockM + mt * 16 + lm + v;
            if (gm < M) out[(size_t)gm * NCOL + gn] = a[v] + bv;
        }
    }
}

// ---------------------------------------------------------------------------
// attention logit: one wave per (edge, head)
// logit[e,h] = sum_c leaky(ep + xl[j] + xr[i], 0.2) * att[h,c]
// ---------------------------------------------------------------------------
template <int Hh, int Cc>
__global__ __launch_bounds__(256) void k_edge_logit(
    const float* __restrict__ ep, const float* __restrict__ xl,
    const float* __restrict__ xr, const int* __restrict__ ei,
    const float* __restrict__ att, float* __restrict__ logits) {
    const int lane = threadIdx.x & 31;
    int gw = blockIdx.x * 8 + (threadIdx.x >> 5);
    if (gw >= ETOT * Hh) return;
    int e = gw / Hh, h = gw % Hh;
    int i = (e < EE) ? ei[e]      : e - EE;
    int j = (e < EE) ? ei[EE + e] : e - EE;
    const int HC = Hh * Cc;
    int c0 = lane * 2;
    f32x2 pe = *(const f32x2*)(ep + (size_t)e * HC + h * Cc + c0);
    f32x2 pl = *(const f32x2*)(xl + (size_t)j * HC + h * Cc + c0);
    f32x2 pr = *(const f32x2*)(xr + (size_t)i * HC + h * Cc + c0);
    f32x2 pa = *(const f32x2*)(att + h * Cc + c0);
    float m0 = lrelu(pe.x + pl.x + pr.x, NEG_GATV2);
    float m1 = lrelu(pe.y + pl.y + pr.y, NEG_GATV2);
    float part = m0 * pa.x + m1 * pa.y;
#pragma unroll
    for (int o = 16; o > 0; o >>= 1) part += __shfl_xor(part, o);
    if (lane == 0) logits[(size_t)e * Hh + h] = part;
}

// segment max over targets (ordered-uint atomic max)
template <int Hh>
__global__ void k_seg_max(const float* __restrict__ logits, const int* __restrict__ ei,
                          unsigned* __restrict__ lmax) {
    int t = blockIdx.x * blockDim.x + threadIdx.x;
    if (t >= ETOT * Hh) return;
    int e = t / Hh, h = t % Hh;
    int i = (e < EE) ? ei[e] : e - EE;
    atomicMax(&lmax[(size_t)i * Hh + h], encf(logits[t]));
}

// ex = exp(logit - max); accumulate denominator
template <int Hh>
__global__ void k_exp_den(float* __restrict__ logits, const int* __restrict__ ei,
                          const unsigned* __restrict__ lmax, float* __restrict__ den) {
    int t = blockIdx.x * blockDim.x + threadIdx.x;
    if (t >= ETOT * Hh) return;
    int e = t / Hh, h = t % Hh;
    int i = (e < EE) ? ei[e] : e - EE;
    float ex = __expf(logits[t] - decf(lmax[(size_t)i * Hh + h]));
    logits[t] = ex;
    atomicAdd(&den[(size_t)i * Hh + h], ex);
}

// out[i,h,:] += alpha * xl[j,h,:]; one wave per (edge, head)
template <int Hh, int Cc>
__global__ __launch_bounds__(256) void k_aggregate(
    const float* __restrict__ exb, const float* __restrict__ den,
    const float* __restrict__ xl, const int* __restrict__ ei,
    float* __restrict__ out) {
    const int lane = threadIdx.x & 31;
    int gw = blockIdx.x * 8 + (threadIdx.x >> 5);
    if (gw >= ETOT * Hh) return;
    int e = gw / Hh, h = gw % Hh;
    int i = (e < EE) ? ei[e]      : e - EE;
    int j = (e < EE) ? ei[EE + e] : e - EE;
    const int HC = Hh * Cc;
    float alpha = exb[(size_t)e * Hh + h] / den[(size_t)i * Hh + h];
    int c0 = lane * 2;
    f32x2 pl = *(const f32x2*)(xl + (size_t)j * HC + h * Cc + c0);
    atomicAdd(&out[(size_t)i * HC + h * Cc + c0 + 0], alpha * pl.x);
    atomicAdd(&out[(size_t)i * HC + h * Cc + c0 + 1], alpha * pl.y);
}

// h = leaky(h + bias, 0.01) in place
__global__ void k_bias_act(float* __restrict__ h, const float* __restrict__ bias,
                           int hc, size_t n) {
    size_t t = (size_t)blockIdx.x * blockDim.x + threadIdx.x;
    if (t >= n) return;
    h[t] = lrelu(h[t] + bias[t % hc], NEG_ACT);
}

// global max pool: g_enc[batch[n], c] = max(h2[n, c])
__global__ void k_pool(const float* __restrict__ h2, const int* __restrict__ batch,
                       unsigned* __restrict__ genc) {
    size_t t = (size_t)blockIdx.x * blockDim.x + threadIdx.x;
    if (t >= (size_t)NN * CC) return;
    int nd = (int)(t >> 6), c = (int)(t & 63);
    atomicMax(&genc[(size_t)batch[nd] * CC + c], encf(h2[t]));
}

// LayerNorm + classifier: one block per graph
__global__ __launch_bounds__(256) void k_final(
    const unsigned* __restrict__ genc, const float* __restrict__ ln_g,
    const float* __restrict__ ln_b, const float* __restrict__ clf_W,
    const float* __restrict__ clf_b, float* __restrict__ out) {
    __shared__ float s_val[CC];
    __shared__ float s_mu, s_rstd;
    int g = blockIdx.x, tid = threadIdx.x;
    if (tid < CC) s_val[tid] = decf(genc[(size_t)g * CC + tid]);
    __syncthreads();
    if (tid == 0) {
        float mu = 0.f;
        for (int k = 0; k < CC; ++k) mu += s_val[k];
        mu /= CC;
        float var = 0.f;
        for (int k = 0; k < CC; ++k) { float d = s_val[k] - mu; var += d * d; }
        var /= CC;
        s_mu = mu; s_rstd = rsqrtf(var + 1e-5f);
    }
    __syncthreads();
    if (tid < CC) {
        float gn = (s_val[tid] - s_mu) * s_rstd * ln_g[tid] + ln_b[tid];
        s_val[tid] = gn;
        out[(size_t)GG * NC_ + (size_t)g * CC + tid] = gn;   // gn output
    }
    __syncthreads();
    for (int c = tid; c < NC_; c += 256) {
        float acc = clf_b[c];
        for (int k = 0; k < CC; ++k) acc += s_val[k] * clf_W[k * NC_ + c];
        out[(size_t)g * NC_ + c] = acc;                      // cls_score output
    }
}

// ---------------------------------------------------------------------------
// host launcher
// ---------------------------------------------------------------------------
static inline int cdiv(long long a, long long b) { return (int)((a + b - 1) / b); }

extern "C" void kernel_launch(void* const* d_in, const int* in_sizes, int n_in,
                              void* d_out, int out_size, void* d_ws, size_t ws_size,
                              hipStream_t stream) {
    const float* x     = (const float*)d_in[0];
    const int*   ei    = (const int*)d_in[1];
    const float* ea    = (const float*)d_in[2];
    const int*   batch = (const int*)d_in[3];
    const float* W1l = (const float*)d_in[4];  const float* b1l = (const float*)d_in[5];
    const float* W1r = (const float*)d_in[6];  const float* b1r = (const float*)d_in[7];
    const float* W1e = (const float*)d_in[8];
    const float* att1  = (const float*)d_in[9];
    const float* bias1 = (const float*)d_in[10];
    const float* W2l = (const float*)d_in[11]; const float* b2l = (const float*)d_in[12];
    const float* W2r = (const float*)d_in[13]; const float* b2r = (const float*)d_in[14];
    const float* W2e = (const float*)d_in[15];
    const float* att2  = (const float*)d_in[16];
    const float* bias2 = (const float*)d_in[17];
    const float* ln_g = (const float*)d_in[18]; const float* ln_b = (const float*)d_in[19];
    const float* clfW = (const float*)d_in[20]; const float* clfb = (const float*)d_in[21];
    float* out = (float*)d_out;

    // workspace layout (floats)
    float* ws = (float*)d_ws;
    const size_t NEL    = (size_t)NN * EDIM_;
    float* lattr = ws;                                // N*EDIM (sum -> mean)
    float* cnt   = lattr + NEL;                       // N
    float* xl1   = cnt + NN;                          // N*256
    float* xr1   = xl1 + (size_t)NN * 256;            // N*256
    float* h1    = xr1 + (size_t)NN * 256;            // N*256 (agg -> act)
    float* xl2   = h1 + (size_t)NN * 256;             // N*64
    float* xr2   = xl2 + (size_t)NN * 64;             // N*64
    float* h2    = xr2 + (size_t)NN * 64;             // N*64
    unsigned* lmax1 = (unsigned*)(h2 + (size_t)NN * 64);  // N*4
    float*    den1  = (float*)(lmax1 + (size_t)NN * 4);   // N*4
    unsigned* lmax2 = (unsigned*)(den1 + (size_t)NN * 4); // N
    float*    den2  = (float*)(lmax2 + NN);               // N
    float* lg1  = den2 + NN;                          // Etot*4
    float* lg2  = lg1 + (size_t)ETOT * 4;             // Etot
    unsigned* genc = (unsigned*)(lg2 + ETOT);         // G*64
    float* ep   = (float*)(genc + GG * CC);           // Etot*256 (reused layer2)

    const int TB = 256;

    // zero init: loop sums + counts | h1 | h2..lmax..den (contiguous 74N) | genc
    k_fill_u32<<<4096, TB, 0, stream>>>((unsigned*)lattr, 0u, NEL + NN);
    k_fill_u32<<<4096, TB, 0, stream>>>((unsigned*)h1, 0u, (size_t)NN * 256);
    k_fill_u32<<<4096, TB, 0, stream>>>((unsigned*)h2, 0u, (size_t)NN * 74);
    k_fill_u32<<<16,   TB, 0, stream>>>(genc, 0u, (size_t)GG * CC);

    // self-loop attr = scatter-mean(edge_attr, col)
    k_scatter_attr<<<cdiv((long long)EE * EDIM_, TB), TB, 0, stream>>>(ea, ei, lattr, cnt);
    k_loop_div<<<cdiv((long long)NN * EDIM_, TB), TB, 0, stream>>>(lattr, cnt);

    // ---------------- layer 1 ----------------
    {
        dim3 gNode(256 / 64, cdiv(NN, 64));
        k_gemm_bf16<512, 256><<<gNode, TB, 0, stream>>>(x, x, NN, NN, W1l, b1l, xl1);
        k_gemm_bf16<512, 256><<<gNode, TB, 0, stream>>>(x, x, NN, NN, W1r, b1r, xr1);
        dim3 gEdge(256 / 64, cdiv(ETOT, 64));
        k_gemm_bf16<384, 256><<<gEdge, TB, 0, stream>>>(ea, lattr, EE, ETOT, W1e, nullptr, ep);
    }
    {
        int wblocks = cdiv((long long)ETOT * HH, 8);
        int tblocks = cdiv((long long)ETOT * HH, TB);
        k_edge_logit<HH, CC><<<wblocks, TB, 0, stream>>>(ep, xl1, xr1, ei, att1, lg1);
        k_seg_max<HH><<<tblocks, TB, 0, stream>>>(lg1, ei, lmax1);
        k_exp_den<HH><<<tblocks, TB, 0, stream>>>(lg1, ei, lmax1, den1);
        k_aggregate<HH, CC><<<wblocks, TB, 0, stream>>>(lg1, den1, xl1, ei, h1);
        k_bias_act<<<cdiv((long long)NN * 256, TB), TB, 0, stream>>>(h1, bias1, 256, (size_t)NN * 256);
    }

    // ---------------- layer 2 ----------------
    {
        dim3 gNode(64 / 64, cdiv(NN, 64));
        k_gemm_bf16<256, 64><<<gNode, TB, 0, stream>>>(h1, h1, NN, NN, W2l, b2l, xl2);
        k_gemm_bf16<256, 64><<<gNode, TB, 0, stream>>>(h1, h1, NN, NN, W2r, b2r, xr2);
        dim3 gEdge(64 / 64, cdiv(ETOT, 64));
        k_gemm_bf16<384, 64><<<gEdge, TB, 0, stream>>>(ea, lattr, EE, ETOT, W2e, nullptr, ep);
    }
    {
        int wblocks = cdiv((long long)ETOT * 1, 8);
        int tblocks = cdiv((long long)ETOT * 1, TB);
        k_edge_logit<1, CC><<<wblocks, TB, 0, stream>>>(ep, xl2, xr2, ei, att2, lg2);
        k_seg_max<1><<<tblocks, TB, 0, stream>>>(lg2, ei, lmax2);
        k_exp_den<1><<<tblocks, TB, 0, stream>>>(lg2, ei, lmax2, den2);
        k_aggregate<1, CC><<<wblocks, TB, 0, stream>>>(lg2, den2, xl2, ei, h2);
        k_bias_act<<<cdiv((long long)NN * 64, TB), TB, 0, stream>>>(h2, bias2, 64, (size_t)NN * 64);
    }

    // pool + LN + classifier
    k_pool<<<cdiv((long long)NN * CC, TB), TB, 0, stream>>>(h2, batch, genc);
    k_final<<<GG, TB, 0, stream>>>(genc, ln_g, ln_b, clfW, clfb, out);
}
